// CATCompressor_59493886984804
// MI455X (gfx1250) — compile-verified, matrix operand order checked by reading, not run on hardware
//
#include <hip/hip_runtime.h>

// ---------------- problem constants (fixed by the harness) ----------------
#define H_DIM  1024
#define NHEADS 16
#define HEADD  64
#define CSZ    32
#define BLK    34             // adaptive + pos + 32 tokens
#define KCH    128
#define BATCH  4
#define TSEQ   (KCH * BLK)    // 4352
#define MTOT   (BATCH * TSEQ) // 17408

typedef __attribute__((ext_vector_type(16))) __bf16         v16bf;
typedef __attribute__((ext_vector_type(8)))  float          v8f;
typedef __attribute__((ext_vector_type(4)))  float          f32x4;
typedef __attribute__((ext_vector_type(4)))  unsigned int   u32x4;
typedef __attribute__((ext_vector_type(4)))  unsigned short u16x4;

union Frag { u32x4 q[2]; v16bf v; };

__device__ __forceinline__ unsigned short f2bf(float f) {
  unsigned u = __builtin_bit_cast(unsigned, f);
  u += 0x7FFFu + ((u >> 16) & 1u);            // round-to-nearest-even
  return (unsigned short)(u >> 16);
}
__device__ __forceinline__ float bf2f(unsigned short h) {
  return __builtin_bit_cast(float, ((unsigned)h) << 16);
}

// ---------------- f32 -> bf16 streaming convert (4 elems / thread) ----------------
__global__ void cvt_bf16_kernel(const float* __restrict__ src,
                                unsigned short* __restrict__ dst, long n4) {
  long i = (long)blockIdx.x * blockDim.x + threadIdx.x;
  if (i >= n4) return;
  f32x4 f = ((const f32x4*)src)[i];
  u16x4 o;
  o[0] = f2bf(f[0]); o[1] = f2bf(f[1]); o[2] = f2bf(f[2]); o[3] = f2bf(f[3]);
  ((u16x4*)dst)[i] = o;
}

// ---------------- embedding assembly ----------------
__global__ void embed_kernel(const int* __restrict__ ids, const float* __restrict__ wte,
                             const float* __restrict__ pos, const float* __restrict__ adaptive,
                             const int* __restrict__ cs_p, float* __restrict__ X) {
  long idx = (long)blockIdx.x * blockDim.x + threadIdx.x;
  if (idx >= (long)MTOT * H_DIM) return;
  int h = (int)(idx % H_DIM);
  int t = (int)((idx / H_DIM) % TSEQ);
  int b = (int)(idx / ((long)H_DIM * TSEQ));
  int kc = t / BLK, p = t % BLK;
  float v;
  if (p == 0)      v = adaptive[(*cs_p) * H_DIM + h];
  else if (p == 1) v = pos[kc * H_DIM + h];
  else {
    int tok = ids[(b * KCH + kc) * CSZ + (p - 2)];
    v = wte[(long)tok * H_DIM + h];
  }
  X[idx] = v;
}

// ---------------- RMSNorm (one block per row), bf16 output for WMMA consumption ----------------
__global__ __launch_bounds__(256) void rmsnorm_kernel(const float* __restrict__ x,
                                                      const float* __restrict__ w,
                                                      unsigned short* __restrict__ out) {
  int row = blockIdx.x;
  int tid = threadIdx.x;
  __shared__ float red[256];
  const float* xr = x + (size_t)row * H_DIM;
  float ss = 0.f;
  for (int h = tid; h < H_DIM; h += 256) { float v = xr[h]; ss += v * v; }
  red[tid] = ss; __syncthreads();
  for (int s = 128; s > 0; s >>= 1) { if (tid < s) red[tid] += red[tid + s]; __syncthreads(); }
  float inv = rsqrtf(red[0] * (1.f / (float)H_DIM) + 1e-6f);
  unsigned short* orow = out + (size_t)row * H_DIM;
  for (int h = tid; h < H_DIM; h += 256) orow[h] = f2bf(xr[h] * inv * w[h]);
}

// ---------------- RoPE, in place on f32 Q or K ----------------
__global__ void rope_kernel(float* __restrict__ buf) {
  long idx = (long)blockIdx.x * blockDim.x + threadIdx.x;
  const long total = (long)MTOT * NHEADS * (HEADD / 2);
  if (idx >= total) return;
  int i    = (int)(idx % 32);
  int head = (int)((idx / 32) % NHEADS);
  long m   = idx / (32 * NHEADS);
  int p    = (int)(m % BLK);
  float ang = (float)p * __expf(-__logf(10000.f) * (float)i * (1.f / 32.f));
  float c = __cosf(ang), s = __sinf(ang);
  float* base = buf + m * H_DIM + head * HEADD + i;
  float x1 = base[0], x2 = base[32];
  base[0]  = x1 * c - x2 * s;
  base[32] = x2 * c + x1 * s;
}

// ---------------- block-local attention, bf16 output (feeds o-proj GEMM) ----------------
__global__ __launch_bounds__(128) void attn_kernel(const float* __restrict__ Q,
                                                   const float* __restrict__ K,
                                                   const float* __restrict__ V,
                                                   unsigned short* __restrict__ O) {
  int idx = blockIdx.x;
  int hd  = idx % NHEADS;
  int kc  = (idx / NHEADS) % KCH;
  int b   = idx / (NHEADS * KCH);
  long row0 = (long)b * TSEQ + (long)kc * BLK;
  int tid = threadIdx.x;
  __shared__ float sq[BLK * HEADD], sk[BLK * HEADD], sv[BLK * HEADD];
  __shared__ float sp[BLK * 36];
  for (int e = tid; e < BLK * HEADD; e += 128) {
    int i = e / HEADD, d = e % HEADD;
    long a = (row0 + i) * H_DIM + hd * HEADD + d;
    sq[e] = Q[a]; sk[e] = K[a]; sv[e] = V[a];
  }
  __syncthreads();
  const float scale = 0.125f;                 // 1/sqrt(64)
  for (int e = tid; e < BLK * BLK; e += 128) {
    int i = e / BLK, j = e % BLK;
    float acc = 0.f;
    #pragma unroll
    for (int d = 0; d < HEADD; ++d) acc += sq[i * HEADD + d] * sk[j * HEADD + d];
    sp[i * 36 + j] = acc * scale;
  }
  __syncthreads();
  if (tid < BLK) {
    float mx = -1e30f;
    for (int j = 0; j < BLK; ++j) mx = fmaxf(mx, sp[tid * 36 + j]);
    float sum = 0.f;
    for (int j = 0; j < BLK; ++j) { float e = __expf(sp[tid * 36 + j] - mx); sp[tid * 36 + j] = e; sum += e; }
    float r = 1.f / sum;
    for (int j = 0; j < BLK; ++j) sp[tid * 36 + j] *= r;
  }
  __syncthreads();
  for (int e = tid; e < BLK * HEADD; e += 128) {
    int i = e / HEADD, d = e % HEADD;
    float acc = 0.f;
    for (int j = 0; j < BLK; ++j) acc += sp[i * 36 + j] * sv[j * HEADD + d];
    O[(row0 + i) * H_DIM + hd * HEADD + d] = f2bf(acc);
  }
}

// ---------------- gather positions 2..33 of each block (bf16 -> bf16) ----------------
__global__ void gather_kernel(const unsigned short* __restrict__ Xn,
                              unsigned short* __restrict__ AF) {
  long idx = (long)blockIdx.x * blockDim.x + threadIdx.x;
  const long total = (long)BATCH * KCH * CSZ * H_DIM;
  if (idx >= total) return;
  int h  = (int)(idx % H_DIM);
  int c  = (int)((idx / H_DIM) % CSZ);
  int kc = (int)((idx / ((long)H_DIM * CSZ)) % KCH);
  int b  = (int)(idx / ((long)H_DIM * CSZ * KCH));
  AF[idx] = Xn[((long)b * TSEQ + kc * BLK + 2 + c) * H_DIM + h];
}

// ---------------- bf16 WMMA GEMM, async-to-LDS double buffering, 64x64 wave tiles ----
// C[M,N] = A[M,K] @ B[N,K]^T   (A,B bf16 row-major)
// mode 0: Cf = acc                      (f32 out)
// mode 1: Cf = acc + extraF[m,n]        (residual, f32 out)
// mode 2: Cb = bf16(silu(acc) * bf16(extraB[m,n]))   (gate x up fusion)
// mode 3: Cb = bf16(acc)                (bf16 out)
// 256 threads = 8 waves; block tile 256x128; waves 4(M)x2(N), each wave 64x64
// -> 16 WMMA per 32-K step from 8 fragment b128 pairs (1.0 LDS b128 per WMMA).
__global__ __launch_bounds__(256) void gemm_wmma_bf16(
    void* __restrict__ Cv, const unsigned short* __restrict__ A,
    const unsigned short* __restrict__ B, const void* __restrict__ extra,
    int M, int N, int K, int mode) {
  (void)M;
  __shared__ __align__(16) unsigned short lsA[2][256 * 40];  // 80B row stride: aligned b128 frags
  __shared__ __align__(16) unsigned short lsB[2][128 * 40];
  const int tid  = threadIdx.x;
  const int m0   = blockIdx.y * 256;
  const int n0   = blockIdx.x * 128;
  const int w    = tid >> 5, lane = tid & 31;
  const int wm   = w & 3;        // M quarter: 64 rows
  const int wn   = w >> 2;       // N half:    64 cols
  const int l16  = lane & 15, hi = lane >> 4;

  // staging coords: each thread async-copies 16B chunks; A rows rs+{0,64,128,192}, B rows rs+{0,64}
  const int rs = tid >> 2;            // 0..63
  const int cc = (tid & 3) * 8;       // half-offset 0,8,16,24

  auto issue_tile = [&](int buf, int k0) {
    #pragma unroll
    for (int p = 0; p < 4; ++p) {
      int r = p * 64 + rs;
      // LDS offset = low 32 bits of the generic address (aperture maps addr[31:0] -> LDS)
      unsigned dA = (unsigned)(size_t)&lsA[buf][r * 40 + cc];
      unsigned long sA = (unsigned long)(const void*)(A + (size_t)(m0 + r) * K + k0 + cc);
      asm volatile("global_load_async_to_lds_b128 %0, %1, off" :: "v"(dA), "v"(sA) : "memory");
    }
    #pragma unroll
    for (int p = 0; p < 2; ++p) {
      int r = p * 64 + rs;
      unsigned dB = (unsigned)(size_t)&lsB[buf][r * 40 + cc];
      unsigned long sB = (unsigned long)(const void*)(B + (size_t)(n0 + r) * K + k0 + cc);
      asm volatile("global_load_async_to_lds_b128 %0, %1, off" :: "v"(dB), "v"(sB) : "memory");
    }
  };

  v8f acc[4][4];
  const v8f vzero = {0.f, 0.f, 0.f, 0.f, 0.f, 0.f, 0.f, 0.f};
  #pragma unroll
  for (int i = 0; i < 4; ++i)
    #pragma unroll
    for (int j = 0; j < 4; ++j) acc[i][j] = vzero;

  issue_tile(0, 0);
  asm volatile("s_wait_asynccnt 0x0" ::: "memory");
  __syncthreads();

  int cur = 0;
  for (int k0 = 0; k0 < K; k0 += 32) {
    if (k0 + 32 < K) issue_tile(cur ^ 1, k0 + 32);   // DMA next tile while computing

    // per-lane fragments: 16-bit A layout -> halves [8*hi,8*hi+8) and [16+8*hi,...)
    Frag aF[4], bF[4];
    #pragma unroll
    for (int i = 0; i < 4; ++i) {
      const unsigned short* rp = &lsA[cur][(wm * 64 + i * 16 + l16) * 40];
      aF[i].q[0] = *(const u32x4*)(rp + 8 * hi);
      aF[i].q[1] = *(const u32x4*)(rp + 16 + 8 * hi);
    }
    #pragma unroll
    for (int j = 0; j < 4; ++j) {
      const unsigned short* rp = &lsB[cur][(wn * 64 + j * 16 + l16) * 40];
      bF[j].q[0] = *(const u32x4*)(rp + 8 * hi);
      bF[j].q[1] = *(const u32x4*)(rp + 16 + 8 * hi);
    }
    #pragma unroll
    for (int i = 0; i < 4; ++i)
      #pragma unroll
      for (int j = 0; j < 4; ++j)
        acc[i][j] = __builtin_amdgcn_wmma_f32_16x16x32_bf16(
            false, aF[i].v, false, bF[j].v, (short)0, acc[i][j], false, false);

    asm volatile("s_wait_asynccnt 0x0" ::: "memory");
    __syncthreads();
    cur ^= 1;
  }

  // epilogue + store: C/D layout -> VGPR r holds M = 8*hi + r, N = lane%16
  float* Cf = (float*)Cv;
  unsigned short* Cb = (unsigned short*)Cv;
  const float* exF = (const float*)extra;
  const unsigned short* exB = (const unsigned short*)extra;
  #pragma unroll
  for (int i = 0; i < 4; ++i) {
    #pragma unroll
    for (int j = 0; j < 4; ++j) {
      int n = n0 + wn * 64 + j * 16 + l16;
      #pragma unroll
      for (int r = 0; r < 8; ++r) {
        int m = m0 + wm * 64 + i * 16 + hi * 8 + r;
        size_t off = (size_t)m * N + n;
        float v = acc[i][j][r];
        if (mode == 0)      Cf[off] = v;
        else if (mode == 1) Cf[off] = v + exF[off];
        else if (mode == 2) {
          float u = bf2f(exB[off]);
          Cb[off] = f2bf((v / (1.f + __expf(-v))) * u);
        } else              Cb[off] = f2bf(v);
      }
    }
  }
}

// ---------------- host orchestration ----------------
extern "C" void kernel_launch(void* const* d_in, const int* in_sizes, int n_in,
                              void* d_out, int out_size, void* d_ws, size_t ws_size,
                              hipStream_t stream) {
  (void)n_in; (void)ws_size;
  const int*   ids    = (const int*)d_in[0];
  const float* wte    = (const float*)d_in[1];
  const float* pos    = (const float*)d_in[2];
  const float* adapt  = (const float*)d_in[3];
  const float* anw    = (const float*)d_in[4];
  const float* q_w    = (const float*)d_in[5];
  const float* k_w    = (const float*)d_in[6];
  const float* v_w    = (const float*)d_in[7];
  const float* o_w    = (const float*)d_in[8];
  const float* mnw    = (const float*)d_in[9];
  const float* gate_w = (const float*)d_in[10];
  const float* up_w   = (const float*)d_in[11];
  const float* down_w = (const float*)d_in[12];
  const float* normw  = (const float*)d_in[13];
  const float* projw  = (const float*)d_in[14];
  const int*   cs_p   = (const int*)d_in[15];
  float* out = (float*)d_out;

  const int  L  = in_sizes[4] / H_DIM;                 // 4 layers
  const long IH = (long)in_sizes[10] / L;              // I*H
  const int  I  = (int)(IH / H_DIM);                   // 2816
  const int  FX = out_size / (BATCH * KCH);            // 2048

  // ---- workspace layout (256B aligned slices) ----
  char* wsp = (char*)d_ws;
  auto alloc = [&](size_t bytes) { char* p = wsp; wsp += (bytes + 255) & ~(size_t)255; return p; };
  float*          X    = (float*)alloc((size_t)MTOT * H_DIM * 4);
  float*          Q    = (float*)alloc((size_t)MTOT * H_DIM * 4);
  float*          Kb   = (float*)alloc((size_t)MTOT * H_DIM * 4);
  float*          V    = (float*)alloc((size_t)MTOT * H_DIM * 4);
  unsigned short* HnB  = (unsigned short*)alloc((size_t)MTOT * H_DIM * 2);
  unsigned short* O    = (unsigned short*)alloc((size_t)MTOT * H_DIM * 2);  // also reused as AF
  unsigned short* Gup  = (unsigned short*)alloc((size_t)MTOT * I * 2);
  unsigned short* G    = (unsigned short*)alloc((size_t)MTOT * I * 2);
  unsigned short* qwB  = (unsigned short*)alloc((size_t)in_sizes[5]  * 2);
  unsigned short* kwB  = (unsigned short*)alloc((size_t)in_sizes[6]  * 2);
  unsigned short* vwB  = (unsigned short*)alloc((size_t)in_sizes[7]  * 2);
  unsigned short* owB  = (unsigned short*)alloc((size_t)in_sizes[8]  * 2);
  unsigned short* gwB  = (unsigned short*)alloc((size_t)in_sizes[10] * 2);
  unsigned short* uwB  = (unsigned short*)alloc((size_t)in_sizes[11] * 2);
  unsigned short* dwB  = (unsigned short*)alloc((size_t)in_sizes[12] * 2);
  unsigned short* pwB  = (unsigned short*)alloc((size_t)in_sizes[14] * 2);
  unsigned short* AF   = O;

  dim3 b256(256);
  auto cvt = [&](const float* s, unsigned short* d, long n) {
    long n4 = n / 4;
    cvt_bf16_kernel<<<(unsigned)((n4 + 255) / 256), b256, 0, stream>>>(s, d, n4);
  };

  // one-shot bf16 conversion of all GEMM weights (halves repeated L2 reads in GEMMs)
  cvt(q_w,    qwB, in_sizes[5]);
  cvt(k_w,    kwB, in_sizes[6]);
  cvt(v_w,    vwB, in_sizes[7]);
  cvt(o_w,    owB, in_sizes[8]);
  cvt(gate_w, gwB, in_sizes[10]);
  cvt(up_w,   uwB, in_sizes[11]);
  cvt(down_w, dwB, in_sizes[12]);
  cvt(projw,  pwB, in_sizes[14]);

  const long elemsXH = (long)MTOT * H_DIM;
  embed_kernel<<<(unsigned)((elemsXH + 255) / 256), b256, 0, stream>>>(ids, wte, pos, adapt, cs_p, X);

  const long ropeTot = (long)MTOT * NHEADS * 32;
  const unsigned ropeGrid = (unsigned)((ropeTot + 255) / 256);

  for (int l = 0; l < L; ++l) {
    rmsnorm_kernel<<<MTOT, b256, 0, stream>>>(X, anw + (size_t)l * H_DIM, HnB);

    dim3 gSq(H_DIM / 128, MTOT / 256);
    gemm_wmma_bf16<<<gSq, b256, 0, stream>>>(Q,  HnB, qwB + (size_t)l * H_DIM * H_DIM, nullptr, MTOT, H_DIM, H_DIM, 0);
    gemm_wmma_bf16<<<gSq, b256, 0, stream>>>(Kb, HnB, kwB + (size_t)l * H_DIM * H_DIM, nullptr, MTOT, H_DIM, H_DIM, 0);
    gemm_wmma_bf16<<<gSq, b256, 0, stream>>>(V,  HnB, vwB + (size_t)l * H_DIM * H_DIM, nullptr, MTOT, H_DIM, H_DIM, 0);

    rope_kernel<<<ropeGrid, b256, 0, stream>>>(Q);
    rope_kernel<<<ropeGrid, b256, 0, stream>>>(Kb);

    attn_kernel<<<BATCH * KCH * NHEADS, 128, 0, stream>>>(Q, Kb, V, O);

    gemm_wmma_bf16<<<gSq, b256, 0, stream>>>(X, O, owB + (size_t)l * H_DIM * H_DIM, X, MTOT, H_DIM, H_DIM, 1);

    rmsnorm_kernel<<<MTOT, b256, 0, stream>>>(X, mnw + (size_t)l * H_DIM, HnB);

    dim3 gUp(I / 128, MTOT / 256);
    gemm_wmma_bf16<<<gUp, b256, 0, stream>>>(Gup, HnB, uwB + (size_t)l * I * H_DIM, nullptr, MTOT, I, H_DIM, 3);
    gemm_wmma_bf16<<<gUp, b256, 0, stream>>>(G,   HnB, gwB + (size_t)l * I * H_DIM, Gup,     MTOT, I, H_DIM, 2);

    dim3 gDn(H_DIM / 128, MTOT / 256);
    gemm_wmma_bf16<<<gDn, b256, 0, stream>>>(X, G, dwB + (size_t)l * H_DIM * I, X, MTOT, H_DIM, I, 1);
  }

  rmsnorm_kernel<<<MTOT, b256, 0, stream>>>(X, normw, HnB);

  const long gTot = (long)BATCH * KCH * CSZ * H_DIM;
  gather_kernel<<<(unsigned)((gTot + 255) / 256), b256, 0, stream>>>(HnB, AF);

  dim3 gF(FX / 128, (BATCH * KCH) / 256);
  gemm_wmma_bf16<<<gF, b256, 0, stream>>>(out, AF, pwB, nullptr, BATCH * KCH, FX, CSZ * H_DIM, 0);
}
